// StructLSTM_45990509806293
// MI455X (gfx1250) — matrix operand order repaired and checked by reading, not compiled, not run on hardware
//
#include <hip/hip_runtime.h>
#include <hip/hip_bf16.h>

typedef _Float16 v16h __attribute__((ext_vector_type(16)));
typedef float    v8f  __attribute__((ext_vector_type(8)));

union F16x16 { v16h v; _Float16 h[16]; };
union F32x8  { v8f  v; float     f[8];  };
union H16x16 { uint4 u[2]; _Float16 h[16]; };
union H16x32 { uint4 u[4]; _Float16 h[32]; };

#define WORD_DIM 300
#define KPAD 320
#define BB 128
#define SS 256
#define H_DIR 384
#define SEMA 512
#define STRU 256
#define G4 1536
#define ROWS (BB*SS)   // 32768

// ---------------- WMMA GEMM, all-f16 operands, fp32 accumulate ----------------
// Requires: M % 64 == 0, N % 128 == 0, K % 64 == 0  (guaranteed by padded buffers)
// A: transA ? (K,M) A[k*lda+m] : (M,K) A[m*lda+k]
// B: transB ? (N,K) B[n*ldb+k] : (K,N) B[k*ldb+n]
#define TM 64
#define TN 128
#define TK 64
#define LDK 72

__launch_bounds__(256)
__global__ void gemm_f16(const _Float16* __restrict__ A, const _Float16* __restrict__ B,
                         const float* __restrict__ bias,
                         float* __restrict__ C, _Float16* __restrict__ C16,
                         int M, int N, int K,
                         long long sAb, long long sBb, long long sCb,
                         int lda, int ldb, int ldc,
                         int transA, int transB, int act)
{
    __shared__ _Float16 As[TM * LDK];
    __shared__ _Float16 Bs[TN * LDK];

    const int tid  = threadIdx.x;
    const int lane = tid & 31;
    const int wid  = tid >> 5;
    const int wm   = wid >> 2;      // 0..1
    const int wn   = wid & 3;       // 0..3
    const int bm   = blockIdx.y * TM;
    const int bn   = blockIdx.x * TN;

    A += (long long)blockIdx.z * sAb;
    B += (long long)blockIdx.z * sBb;

    F32x8 acc[2][2];
#pragma unroll
    for (int i = 0; i < 2; ++i)
#pragma unroll
        for (int j = 0; j < 2; ++j)
#pragma unroll
            for (int r = 0; r < 8; ++r) acc[i][j].f[r] = 0.0f;

    const int mrow   = lane & 15;
    const int aKbase = (lane < 16) ? 0 : 16;   // A frag K base (contiguous k = 16h + e, matches B)
    const int bKbase = (lane < 16) ? 0 : 16;   // B frag K base (contiguous k = 16h + e, matches A)

    // per-thread staging coordinates (uniform pattern, full tiles)
    const int arF = tid >> 2, acF = (tid & 3) * 16;          // A fast path: 32B/thread
    const int akT = tid >> 2, amT = (tid & 3) * 16;          // A transposed
    const int bnF = tid >> 1, bcF = (tid & 1) * 32;          // B NT fast path: 64B/thread
    const int bkT = tid >> 2, bnT = (tid & 3) * 32;          // B NN

    const unsigned ldsA = (unsigned)(uintptr_t)(&As[arF * LDK + acF]);
    const unsigned ldsB = (unsigned)(uintptr_t)(&Bs[bnF * LDK + bcF]);

    for (int k0 = 0; k0 < K; k0 += TK) {
        // ---- stage A tile (TM x TK) ----
        if (!transA) {
            const _Float16* src = A + (long long)(bm + arF) * lda + k0 + acF;
            // CDNA5 async DMA: global -> LDS, ASYNCcnt-tracked, same imm offset on both sides
            asm volatile(
                "global_load_async_to_lds_b128 %0, %1, off\n\t"
                "global_load_async_to_lds_b128 %0, %1, off offset:16"
                :: "v"(ldsA), "v"(src) : "memory");
        } else {
            const uint4* src = (const uint4*)(A + (long long)(k0 + akT) * lda + bm + amT);
            __builtin_prefetch(src + 8, 0, 1);
            H16x16 t; t.u[0] = src[0]; t.u[1] = src[1];
#pragma unroll
            for (int i = 0; i < 16; ++i) As[(amT + i) * LDK + akT] = t.h[i];
        }
        // ---- stage B tile (TN x TK) as [n][k] ----
        if (transB) {
            const _Float16* src = B + (long long)(bn + bnF) * ldb + k0 + bcF;
            asm volatile(
                "global_load_async_to_lds_b128 %0, %1, off\n\t"
                "global_load_async_to_lds_b128 %0, %1, off offset:16\n\t"
                "global_load_async_to_lds_b128 %0, %1, off offset:32\n\t"
                "global_load_async_to_lds_b128 %0, %1, off offset:48"
                :: "v"(ldsB), "v"(src) : "memory");
        } else {
            const uint4* src = (const uint4*)(B + (long long)(k0 + bkT) * ldb + bn + bnT);
            __builtin_prefetch(src + 8, 0, 1);
            H16x32 t;
#pragma unroll
            for (int i = 0; i < 4; ++i) t.u[i] = src[i];
#pragma unroll
            for (int i = 0; i < 32; ++i) Bs[(bnT + i) * LDK + bkT] = t.h[i];
        }
        if (!transA || transB)
            asm volatile("s_wait_asynccnt 0" ::: "memory");
        __syncthreads();

#pragma unroll
        for (int ks = 0; ks < TK; ks += 32) {
            F16x16 af[2], bf[2];
#pragma unroll
            for (int mf = 0; mf < 2; ++mf) {
                int row = wm * 32 + mf * 16 + mrow;
#pragma unroll
                for (int e = 0; e < 16; ++e) {
                    int kk = aKbase + e;
                    af[mf].h[e] = As[row * LDK + ks + kk];
                }
            }
#pragma unroll
            for (int nf = 0; nf < 2; ++nf) {
                int col = wn * 32 + nf * 16 + mrow;
#pragma unroll
                for (int e = 0; e < 16; ++e)
                    bf[nf].h[e] = Bs[col * LDK + ks + bKbase + e];
            }
#pragma unroll
            for (int mf = 0; mf < 2; ++mf)
#pragma unroll
                for (int nf = 0; nf < 2; ++nf)
                    acc[mf][nf].v = __builtin_amdgcn_wmma_f32_16x16x32_f16(
                        false, af[mf].v, false, bf[nf].v,
                        (short)0, acc[mf][nf].v, false, false);
        }
        __syncthreads();
    }

    const int crhi = (lane >> 4) * 8;
    const int ncol = lane & 15;
#pragma unroll
    for (int mf = 0; mf < 2; ++mf)
#pragma unroll
        for (int nf = 0; nf < 2; ++nf)
#pragma unroll
            for (int r = 0; r < 8; ++r) {
                int gm = bm + wm * 32 + mf * 16 + crhi + r;
                int gn = bn + wn * 32 + nf * 16 + ncol;
                float v = acc[mf][nf].f[r];
                if (bias) v += bias[gn];
                if (act == 1) v = tanhf(v);
                long long o = (long long)blockIdx.z * sCb + (long long)gm * ldc + gn;
                if (C)   C[o]   = v;
                if (C16) C16[o] = (_Float16)v;
            }
}

// ---------------- helpers ----------------
__device__ __forceinline__ float sigm(float x) { return 1.0f / (1.0f + expf(-x)); }

__global__ void k_zero(float* p, long long n) {
    long long i = (long long)blockIdx.x * blockDim.x + threadIdx.x;
    if (i < n) p[i] = 0.0f;
}
__global__ void k_zero16(_Float16* p, long long n) {
    long long i = (long long)blockIdx.x * blockDim.x + threadIdx.x;
    if (i < n) p[i] = (_Float16)0.0f;
}

// fp32 (R,C) -> f16 (R,Cp) zero padded
__global__ void k_cvt_pad(const float* __restrict__ src, _Float16* __restrict__ dst,
                          int R, int C, int Cp)
{
    long long i = (long long)blockIdx.x * blockDim.x + threadIdx.x;
    if (i >= (long long)R * Cp) return;
    int r = (int)(i / Cp);
    int c = (int)(i - (long long)r * Cp);
    dst[i] = (c < C) ? (_Float16)src[(long long)r * C + c] : (_Float16)0.0f;
}

__global__ void k_lstm_gate(const float* __restrict__ Gf, const float* __restrict__ Gb,
                            const float* __restrict__ Hmat,
                            float* c_state, _Float16* h16,
                            float* hs_f, float* hs_b, int t)
{
    int idx = blockIdx.x * blockDim.x + threadIdx.x;
    if (idx >= 2 * BB * H_DIR) return;
    int d = idx / (BB * H_DIR);
    int rem = idx - d * (BB * H_DIR);
    int b = rem / H_DIR;
    int h = rem - b * H_DIR;
    int pos = (d == 0) ? t : (SS - 1 - t);
    const float* G = (d == 0) ? Gf : Gb;
    long long grow = ((long long)b * SS + pos) * G4;
    const float* Hm = Hmat + ((long long)d * BB + b) * G4;
    float ig = G[grow + h]             + Hm[h];
    float fg = G[grow + H_DIR + h]     + Hm[H_DIR + h];
    float gg = G[grow + 2 * H_DIR + h] + Hm[2 * H_DIR + h];
    float og = G[grow + 3 * H_DIR + h] + Hm[3 * H_DIR + h];
    float c = c_state[idx];
    c = sigm(fg) * c + sigm(ig) * tanhf(gg);
    float hn = sigm(og) * tanhf(c);
    c_state[idx] = c;
    h16[idx]     = (_Float16)hn;
    float* hs = (d == 0) ? hs_f : hs_b;
    hs[((long long)b * SS + pos) * H_DIR + h] = hn;
}

// Cat16 cols 0..511 = vec_sema ; stru16 separate (256)
__global__ void k_build_cat(const float* __restrict__ hs_f, const float* __restrict__ hs_b,
                            _Float16* __restrict__ Cat16, _Float16* __restrict__ stru16)
{
    long long i = (long long)blockIdx.x * blockDim.x + threadIdx.x;
    if (i >= (long long)ROWS * 768) return;
    int row = (int)(i / 768);
    int col = (int)(i - (long long)row * 768);
    if (col < 512) {
        float v = (col < 256) ? hs_f[(long long)row * H_DIR + col]
                              : hs_b[(long long)row * H_DIR + (col - 256)];
        Cat16[(long long)row * G4 + col] = (_Float16)v;
    } else {
        int c2 = col - 512;
        float v = (c2 < 128) ? hs_f[(long long)row * H_DIR + 256 + c2]
                             : hs_b[(long long)row * H_DIR + 256 + (c2 - 128)];
        stru16[(long long)row * STRU + c2] = (_Float16)v;
    }
}

__global__ void k_f_root(const _Float16* __restrict__ stru16, const float* __restrict__ w_root,
                         const float* __restrict__ mask, float* __restrict__ fr)
{
    int row = blockIdx.x * blockDim.x + threadIdx.x;
    if (row >= ROWS) return;
    float acc = 0.0f;
    const _Float16* s = stru16 + (long long)row * STRU;
    for (int k = 0; k < STRU; ++k) acc += (float)s[k] * w_root[k];
    fr[row] = expf(acc) * mask[row];
}

__global__ void k_f_finalize(float* f, const float* __restrict__ mask) {
    long long i = (long long)blockIdx.x * blockDim.x + threadIdx.x;
    if (i >= (long long)BB * SS * SS) return;
    int b = (int)(i / (SS * SS));
    int rem = (int)(i - (long long)b * SS * SS);
    int s = rem / SS, t = rem - s * SS;
    f[i] = (s == t) ? 0.0f : expf(f[i]) * mask[b * SS + s] * mask[b * SS + t];
}

__global__ void k_colsum(const float* __restrict__ f, float* __restrict__ cs) {
    int i = blockIdx.x * blockDim.x + threadIdx.x;   // (b,t)
    if (i >= BB * SS) return;
    int b = i / SS, t = i - b * SS;
    const float* fb = f + (long long)b * SS * SS;
    float acc = 0.0f;
    for (int s = 0; s < SS; ++s) acc += fb[(long long)s * SS + t];
    cs[i] = acc;
}

__global__ void k_build_LL(const float* __restrict__ f, const float* __restrict__ fr,
                           const float* __restrict__ cs, float* __restrict__ LL)
{
    long long i = (long long)blockIdx.x * blockDim.x + threadIdx.x;
    if (i >= (long long)BB * SS * SS) return;
    int b = (int)(i / (SS * SS));
    int rem = (int)(i - (long long)b * SS * SS);
    int r = rem / SS, c = rem - r * SS;
    float v;
    if (r == 0)      v = fr[b * SS + c];
    else if (r == c) v = cs[b * SS + c];
    else             v = -f[i];
    if (r == c && v == 0.0f) v = 1.0f;   // EPS_MT
    LL[i] = v;
}

// in-place Gauss-Jordan inverse, one 256x256 matrix per workgroup in LDS (CDNA5: 320KB/WGP)
#define NINV 256
#define INVLD 257
__launch_bounds__(256)
__global__ void k_inverse(const float* __restrict__ LL, float* __restrict__ out) {
    extern __shared__ float Ms[];
    __shared__ float pinv_s;
    int tid = threadIdx.x;
    const float* src = LL + (long long)blockIdx.x * NINV * NINV;
    float* dst = out + (long long)blockIdx.x * NINV * NINV;
    for (int idx = tid; idx < NINV * NINV; idx += 256) {
        int i = idx >> 8, j = idx & 255;
        Ms[i * INVLD + j] = src[idx];
    }
    __syncthreads();
    for (int k = 0; k < NINV; ++k) {
        if (tid == 0) pinv_s = 1.0f / Ms[k * INVLD + k];
        __syncthreads();
        float pinv = pinv_s;
        if (tid != k) Ms[k * INVLD + tid] *= pinv;
        __syncthreads();
        if (tid != k) {
            float fac = Ms[tid * INVLD + k];
            for (int j = 0; j < NINV; ++j)
                if (j != k) Ms[tid * INVLD + j] -= fac * Ms[k * INVLD + j];
            Ms[tid * INVLD + k] = -fac * pinv;
        }
        __syncthreads();
        if (tid == 0) Ms[k * INVLD + k] = pinv;
        __syncthreads();
    }
    for (int idx = tid; idx < NINV * NINV; idx += 256) {
        int i = idx >> 8, j = idx & 255;
        dst[idx] = Ms[i * INVLD + j];
    }
}

__global__ void k_d0(const float* __restrict__ fr, const float* __restrict__ LLi,
                     float* __restrict__ d0)
{
    int i = blockIdx.x * blockDim.x + threadIdx.x;  // (b,t)
    if (i >= BB * SS) return;
    int b = i / SS, t = i - b * SS;
    d0[i] = fr[i] * LLi[((long long)b * SS + t) * SS + 0];
}

__global__ void k_attach(const float* __restrict__ f, const float* __restrict__ LLi,
                         _Float16* __restrict__ a16)
{
    long long i = (long long)blockIdx.x * blockDim.x + threadIdx.x;
    if (i >= (long long)BB * SS * SS) return;
    int b = (int)(i / (SS * SS));
    int rem = (int)(i - (long long)b * SS * SS);
    int s = rem / SS, t = rem - s * SS;
    const float* Lb = LLi + (long long)b * SS * SS;
    float t1 = (t == 0) ? 0.0f : Lb[(long long)t * SS + t];
    float t2 = (s == 0) ? 0.0f : Lb[(long long)t * SS + s];
    a16[i] = (_Float16)(f[i] * (t1 - t2));
}

__global__ void k_add_outer(_Float16* __restrict__ Cat16, const float* __restrict__ d0,
                            const float* __restrict__ root_emb)
{
    long long i = (long long)blockIdx.x * blockDim.x + threadIdx.x;
    if (i >= (long long)ROWS * SEMA) return;
    int row = (int)(i / SEMA);
    int n = (int)(i - (long long)row * SEMA);
    long long o = (long long)row * G4 + 512 + n;
    Cat16[o] = (_Float16)((float)Cat16[o] + d0[row] * root_emb[n]);
}

__global__ void k_mask_max(float* __restrict__ out, const float* __restrict__ mask,
                           float* __restrict__ emb)
{
    int i = blockIdx.x * blockDim.x + threadIdx.x;  // (b,n)
    if (i >= BB * SEMA) return;
    int b = i / SEMA, n = i - b * SEMA;
    float mx = -3.4e38f;
    for (int s = 0; s < SS; ++s) {
        long long o = ((long long)b * SS + s) * SEMA + n;
        float v = out[o] * mask[b * SS + s];
        out[o] = v;
        if (v > mx) mx = v;
    }
    emb[(long long)b * SEMA + n] = mx;
}

// ---------------- launch ----------------
static inline dim3 gemm_grid(int M, int N, int batch) {
    return dim3((unsigned)(N / TN), (unsigned)(M / TM), (unsigned)batch);
}
static inline unsigned cdiv(long long n, int b) { return (unsigned)((n + b - 1) / b); }

extern "C" void kernel_launch(void* const* d_in, const int* in_sizes, int n_in,
                              void* d_out, int out_size, void* d_ws, size_t ws_size,
                              hipStream_t stream)
{
    const float* X      = (const float*)d_in[0];
    const float* mask   = (const float*)d_in[1];
    const float* w_ih_f = (const float*)d_in[2];
    const float* w_hh_f = (const float*)d_in[3];
    const float* b_f    = (const float*)d_in[4];
    const float* w_ih_b = (const float*)d_in[5];
    const float* w_hh_b = (const float*)d_in[6];
    const float* b_b    = (const float*)d_in[7];
    const float* w_tp   = (const float*)d_in[8];
    const float* b_tp   = (const float*)d_in[9];
    const float* w_tc   = (const float*)d_in[10];
    const float* b_tc   = (const float*)d_in[11];
    const float* w_a    = (const float*)d_in[12];
    const float* w_root = (const float*)d_in[13];
    const float* root_e = (const float*)d_in[14];
    const float* w_r    = (const float*)d_in[15];
    const float* b_r    = (const float*)d_in[16];

    float* emb_out = (float*)d_out;                 // 128*512
    float* out_buf = (float*)d_out + BB * SEMA;     // 128*256*512

    char* base = (char*)d_ws;
    size_t off = 0;
    auto allocF = [&](size_t elems) {
        float* p = (float*)(base + off); off += ((elems * 4 + 255) & ~(size_t)255); return p;
    };
    auto allocH = [&](size_t elems) {
        _Float16* p = (_Float16*)(base + off); off += ((elems * 2 + 255) & ~(size_t)255); return p;
    };
    // fp32 buffers
    float* G_f   = allocF((size_t)ROWS * G4);
    float* G_b   = allocF((size_t)ROWS * G4);
    float* Hmat  = allocF((size_t)2 * BB * G4);
    float* c_st  = allocF((size_t)2 * BB * H_DIR);
    float* hs_f  = allocF((size_t)ROWS * H_DIR);
    float* hs_b  = allocF((size_t)ROWS * H_DIR);
    float* fmat  = allocF((size_t)BB * SS * SS);
    float* LL    = allocF((size_t)BB * SS * SS);
    float* LLi   = allocF((size_t)BB * SS * SS);
    float* fr    = allocF((size_t)ROWS);
    float* cs    = allocF((size_t)ROWS);
    float* d0    = allocF((size_t)ROWS);
    // f16 buffers
    _Float16* X16    = allocH((size_t)ROWS * KPAD);
    _Float16* wihf16 = allocH((size_t)G4 * KPAD);
    _Float16* wihb16 = allocH((size_t)G4 * KPAD);
    _Float16* whh16  = allocH((size_t)2 * G4 * H_DIR);
    _Float16* wtp16  = allocH((size_t)STRU * STRU);
    _Float16* wtc16  = allocH((size_t)STRU * STRU);
    _Float16* wa16   = allocH((size_t)STRU * STRU);
    _Float16* wr16   = allocH((size_t)SEMA * G4);
    _Float16* h16    = allocH((size_t)2 * BB * H_DIR);
    _Float16* stru16 = allocH((size_t)ROWS * STRU);
    _Float16* tp16   = allocH((size_t)ROWS * STRU);
    _Float16* tc16   = allocH((size_t)ROWS * STRU);
    _Float16* u16    = allocH((size_t)ROWS * STRU);
    _Float16* a16    = allocH((size_t)BB * SS * SS);
    _Float16* Cat16  = allocH((size_t)ROWS * G4);

    // ---- convert / pad operands to f16 ----
    k_cvt_pad<<<cdiv((long long)ROWS * KPAD, 256), 256, 0, stream>>>(X, X16, ROWS, WORD_DIM, KPAD);
    k_cvt_pad<<<cdiv((long long)G4 * KPAD, 256), 256, 0, stream>>>(w_ih_f, wihf16, G4, WORD_DIM, KPAD);
    k_cvt_pad<<<cdiv((long long)G4 * KPAD, 256), 256, 0, stream>>>(w_ih_b, wihb16, G4, WORD_DIM, KPAD);
    k_cvt_pad<<<cdiv((long long)G4 * H_DIR, 256), 256, 0, stream>>>(w_hh_f, whh16, G4, H_DIR, H_DIR);
    k_cvt_pad<<<cdiv((long long)G4 * H_DIR, 256), 256, 0, stream>>>(w_hh_b, whh16 + (size_t)G4 * H_DIR, G4, H_DIR, H_DIR);
    k_cvt_pad<<<cdiv((long long)STRU * STRU, 256), 256, 0, stream>>>(w_tp, wtp16, STRU, STRU, STRU);
    k_cvt_pad<<<cdiv((long long)STRU * STRU, 256), 256, 0, stream>>>(w_tc, wtc16, STRU, STRU, STRU);
    k_cvt_pad<<<cdiv((long long)STRU * STRU, 256), 256, 0, stream>>>(w_a, wa16, STRU, STRU, STRU);
    k_cvt_pad<<<cdiv((long long)SEMA * G4, 256), 256, 0, stream>>>(w_r, wr16, SEMA, G4, G4);

    // ---- init recurrent state ----
    k_zero  <<<cdiv(2LL * BB * H_DIR, 256), 256, 0, stream>>>(c_st, 2LL * BB * H_DIR);
    k_zero16<<<cdiv(2LL * BB * H_DIR, 256), 256, 0, stream>>>(h16, 2LL * BB * H_DIR);

    // ---- input projections: G = X @ w_ih^T + b  (M=32768,N=1536,K=320) ----
    gemm_f16<<<gemm_grid(ROWS, G4, 1), 256, 0, stream>>>(
        X16, wihf16, b_f, G_f, nullptr, ROWS, G4, KPAD, 0, 0, 0, KPAD, KPAD, G4, 0, 1, 0);
    gemm_f16<<<gemm_grid(ROWS, G4, 1), 256, 0, stream>>>(
        X16, wihb16, b_b, G_b, nullptr, ROWS, G4, KPAD, 0, 0, 0, KPAD, KPAD, G4, 0, 1, 0);

    // ---- LSTM recurrence (2 dirs batched per step) ----
    for (int t = 0; t < SS; ++t) {
        gemm_f16<<<gemm_grid(BB, G4, 2), 256, 0, stream>>>(
            h16, whh16, nullptr, Hmat, nullptr, BB, G4, H_DIR,
            (long long)BB * H_DIR, (long long)G4 * H_DIR, (long long)BB * G4,
            H_DIR, H_DIR, G4, 0, 1, 0);
        k_lstm_gate<<<cdiv(2LL * BB * H_DIR, 256), 256, 0, stream>>>(
            G_f, G_b, Hmat, c_st, h16, hs_f, hs_b, t);
    }

    // ---- split: vec_sema -> Cat16[:,0:512], vec_stru -> stru16 ----
    k_build_cat<<<cdiv((long long)ROWS * 768, 256), 256, 0, stream>>>(hs_f, hs_b, Cat16, stru16);

    // ---- tp / tc (tanh) -> f16 ----
    gemm_f16<<<gemm_grid(ROWS, STRU, 1), 256, 0, stream>>>(
        stru16, wtp16, b_tp, nullptr, tp16, ROWS, STRU, STRU, 0, 0, 0, STRU, STRU, STRU, 0, 1, 1);
    gemm_f16<<<gemm_grid(ROWS, STRU, 1), 256, 0, stream>>>(
        stru16, wtc16, b_tc, nullptr, tc16, ROWS, STRU, STRU, 0, 0, 0, STRU, STRU, STRU, 0, 1, 1);

    // ---- u = tp @ w_a (NN) -> f16 ----
    gemm_f16<<<gemm_grid(ROWS, STRU, 1), 256, 0, stream>>>(
        tp16, wa16, nullptr, nullptr, u16, ROWS, STRU, STRU, 0, 0, 0, STRU, STRU, STRU, 0, 0, 0);

    // ---- f_raw = u @ tc^T per batch -> fp32 ----
    gemm_f16<<<gemm_grid(SS, SS, BB), 256, 0, stream>>>(
        u16, tc16, nullptr, fmat, nullptr, SS, SS, STRU,
        (long long)SS * STRU, (long long)SS * STRU, (long long)SS * SS,
        STRU, STRU, SS, 0, 1, 0);

    // ---- matrix-tree prep ----
    k_f_root<<<cdiv(ROWS, 256), 256, 0, stream>>>(stru16, w_root, mask, fr);
    {
        long long n = (long long)BB * SS * SS;
        k_f_finalize<<<cdiv(n, 256), 256, 0, stream>>>(fmat, mask);
        k_colsum<<<cdiv((long long)BB * SS, 256), 256, 0, stream>>>(fmat, cs);
        k_build_LL<<<cdiv(n, 256), 256, 0, stream>>>(fmat, fr, cs, LL);
    }

    // ---- batched 256x256 inverse in LDS ----
    k_inverse<<<dim3(BB), 256, NINV * INVLD * sizeof(float), stream>>>(LL, LLi);

    k_d0<<<cdiv((long long)BB * SS, 256), 256, 0, stream>>>(fr, LLi, d0);
    k_attach<<<cdiv((long long)BB * SS * SS, 256), 256, 0, stream>>>(fmat, LLi, a16);

    // ---- c = a @ vec_sema -> Cat16[:,1024:1536] ----
    gemm_f16<<<gemm_grid(SS, SEMA, BB), 256, 0, stream>>>(
        a16, Cat16, nullptr, nullptr, Cat16 + 1024, SS, SEMA, STRU,
        (long long)SS * SS, (long long)SS * G4, (long long)SS * G4,
        SS, G4, G4, 0, 0, 0);
    // ---- p = a^T @ vec_sema -> Cat16[:,512:1024] ----
    gemm_f16<<<gemm_grid(SS, SEMA, BB), 256, 0, stream>>>(
        a16, Cat16, nullptr, nullptr, Cat16 + 512, SS, SEMA, STRU,
        (long long)SS * SS, (long long)SS * G4, (long long)SS * G4,
        SS, G4, G4, 1, 0, 0);
    k_add_outer<<<cdiv((long long)ROWS * SEMA, 256), 256, 0, stream>>>(Cat16, d0, root_e);

    // ---- output = tanh(Cat @ w_r^T + b_r) ----
    gemm_f16<<<gemm_grid(ROWS, SEMA, 1), 256, 0, stream>>>(
        Cat16, wr16, b_r, out_buf, nullptr, ROWS, SEMA, G4, 0, 0, 0, G4, G4, SEMA, 0, 1, 1);

    // ---- mask + max-reduce -> emb ----
    k_mask_max<<<cdiv((long long)BB * SEMA, 256), 256, 0, stream>>>(out_buf, mask, emb_out);

    (void)in_sizes; (void)n_in; (void)out_size; (void)ws_size;
}
